// CoreAttention_34875134444341
// MI455X (gfx1250) — matrix operand order repaired and checked by reading, not compiled, not run on hardware
//
#include <hip/hip_runtime.h>

// ---------------------------------------------------------------------------
// CDNA5 (gfx1250) MLA + GQA causal attention, bf16 WMMA pipeline with
// async global->LDS tile staging (ASYNCcnt), offset-immediate chunking,
// and double-buffered LDS in both GEMM and attention.
// ---------------------------------------------------------------------------

typedef __attribute__((ext_vector_type(16))) __bf16 v16bf;
typedef __attribute__((ext_vector_type(8)))  float  v8f;

static constexpr int kB    = 4;
static constexpr int kS    = 2048;
static constexpr int kD    = 2048;
static constexpr int kH    = 16;
static constexpr int kHKV  = 4;
static constexpr int kDK   = 128;      // head dim
static constexpr int kGrp  = kH / kHKV;
static constexpr int kKVW  = kHKV * kDK;   // 512 = kv row width

__device__ __forceinline__ v8f wmma_bf16(v16bf a, v16bf b, v8f c) {
  return __builtin_amdgcn_wmma_f32_16x16x32_bf16(false, a, false, b,
                                                 (short)0, c, false, false);
}

// Async global -> LDS, 16 bytes, IOFFSET applies to both addresses (CDNA5).
template <int OFF>
__device__ __forceinline__ void async_b128(uint32_t lds_addr, uint64_t gaddr) {
  asm volatile("global_load_async_to_lds_b128 %0, %1, off offset:%2"
               :: "v"(lds_addr), "v"(gaddr), "i"(OFF) : "memory");
}
__device__ __forceinline__ uint32_t lds_off(const void* p) {
  return (uint32_t)(uintptr_t)p;  // low 32 bits of generic ptr = LDS offset
}
__device__ __forceinline__ void wait_async0() {
  asm volatile("s_wait_asynccnt 0" ::: "memory");
}

// A-operand fragment (16x32 bf16, row major source, stride ld).
// lane<16 -> M=lane, K in {0..15}; lane>=16 -> K in {16..31} (contiguous, matches B)
__device__ __forceinline__ v16bf ld_frag_a(const __bf16* __restrict__ p, int ld) {
  const int lane = threadIdx.x & 31;
  const int half = lane >> 4;
  const int r    = lane & 15;
  v16bf f;
#pragma unroll
  for (int i = 0; i < 8; ++i) {
    const int kb = (half << 4) + (i << 1);
    f[2 * i]     = p[r * ld + kb];
    f[2 * i + 1] = p[r * ld + kb + 1];
  }
  return f;
}

// B-operand fragment (32x16 bf16, [K][N] row-major source, stride ld).
// lanes 0-15 hold K=0..15 (contiguous), lanes 16-31 hold K=16..31; N = lane&15.
__device__ __forceinline__ v16bf ld_frag_b(const __bf16* __restrict__ p, int ld) {
  const int lane = threadIdx.x & 31;
  const int half = lane >> 4;
  const int c    = lane & 15;
  v16bf f;
#pragma unroll
  for (int i = 0; i < 8; ++i) {
    const int kb = (half << 4) + (i << 1);
    f[2 * i]     = p[kb * ld + c];
    f[2 * i + 1] = p[(kb + 1) * ld + c];
  }
  return f;
}

// ---------------------------------------------------------------------------
// fp32 -> bf16 conversion
// ---------------------------------------------------------------------------
__global__ void cvt_f32_bf16(const float* __restrict__ src,
                             __bf16* __restrict__ dst, int n) {
  int i = blockIdx.x * blockDim.x + threadIdx.x;
  const int stride = gridDim.x * blockDim.x;
  for (; i < n; i += stride) dst[i] = (__bf16)src[i];
}

// ---------------------------------------------------------------------------
// Tiled bf16 GEMM: C[M,N] = A[M,K] @ B[K,N].
// Block tile 128x256, 8 waves (2x4), wave tile 64x64 (4x4 fragments).
// K-step 64 through double-buffered LDS (96 KB) filled by async loads;
// one barrier + one per-wave asynccnt wait per K-step, 32 WMMA between.
// ---------------------------------------------------------------------------
template <typename OutT>
__global__ __launch_bounds__(256) void gemm_bf16(
    const __bf16* __restrict__ A, const __bf16* __restrict__ Bm,
    OutT* __restrict__ Cp, int M, int N, int K) {
  __shared__ alignas(16) __bf16 sA[2][128 * 64];
  __shared__ alignas(16) __bf16 sB[2][64 * 256];

  const int tid  = threadIdx.x;
  const int wave = tid >> 5;
  const int lane = tid & 31;
  const int half = lane >> 4;
  const int cn   = lane & 15;
  const int wm   = wave >> 2;  // 0..1  (64-row strips)
  const int wn   = wave & 3;   // 0..3  (64-col strips)
  const int bm   = blockIdx.y * 128;
  const int bn   = blockIdx.x * 256;

  v8f acc[4][4];
#pragma unroll
  for (int mi = 0; mi < 4; ++mi)
#pragma unroll
    for (int ni = 0; ni < 4; ++ni) acc[mi][ni] = {};

  // Per-thread tile-load geometry (A: 128x64, 32 bf16/thread; B: 64x256,
  // 64 bf16/thread), equal byte strides in LDS and global per chunk.
  const int ar  = tid >> 1;
  const int ac0 = (tid & 1) << 5;   // 32 elems = 4 b128
  const int br  = tid >> 2;
  const int bc0 = (tid & 3) << 6;   // 64 elems = 8 b128

  auto issue_tile = [&](int k0, int buf) {
    {
      const uint64_t g = (uint64_t)(uintptr_t)(A + (size_t)(bm + ar) * K + k0 + ac0);
      const uint32_t l = lds_off(&sA[buf][ar * 64 + ac0]);
      async_b128<0>(l, g);  async_b128<16>(l, g);
      async_b128<32>(l, g); async_b128<48>(l, g);
    }
    {
      const uint64_t g = (uint64_t)(uintptr_t)(Bm + (size_t)(k0 + br) * N + bn + bc0);
      const uint32_t l = lds_off(&sB[buf][br * 256 + bc0]);
      async_b128<0>(l, g);   async_b128<16>(l, g);
      async_b128<32>(l, g);  async_b128<48>(l, g);
      async_b128<64>(l, g);  async_b128<80>(l, g);
      async_b128<96>(l, g);  async_b128<112>(l, g);
    }
  };

  issue_tile(0, 0);
  int cur = 0;
  for (int k0 = 0; k0 < K; k0 += 64) {
    wait_async0();      // my async writes into sX[cur] are done
    __syncthreads();    // everyone's are done; prior reads of sX[cur^1] retired
    if (k0 + 64 < K) issue_tile(k0 + 64, cur ^ 1);

#pragma unroll
    for (int kk = 0; kk < 64; kk += 32) {
      v16bf af[4];
#pragma unroll
      for (int mi = 0; mi < 4; ++mi)
        af[mi] = ld_frag_a(&sA[cur][(wm * 64 + mi * 16) * 64 + kk], 64);
#pragma unroll
      for (int ni = 0; ni < 4; ++ni) {
        v16bf bf = ld_frag_b(&sB[cur][kk * 256 + wn * 64 + ni * 16], 256);
#pragma unroll
        for (int mi = 0; mi < 4; ++mi)
          acc[mi][ni] = wmma_bf16(af[mi], bf, acc[mi][ni]);
      }
    }
    cur ^= 1;
  }

#pragma unroll
  for (int mi = 0; mi < 4; ++mi)
#pragma unroll
    for (int ni = 0; ni < 4; ++ni)
#pragma unroll
      for (int jj = 0; jj < 8; ++jj) {
        const int row = bm + wm * 64 + mi * 16 + jj + 8 * half;
        const int col = bn + wn * 64 + ni * 16 + cn;
        Cp[(size_t)row * N + col] = (OutT)acc[mi][ni][jj];
      }
}

// ---------------------------------------------------------------------------
// Flash-style causal attention.  Grid: (S/64, B*H).  128 threads = 4 waves;
// each wave owns a 16-row Q strip.  K/V tiles (64x128) double-buffered in
// LDS via async loads; P is round-tripped through LDS to re-fragment as an
// A operand.  One barrier per KV tile; scores use paired independent
// accumulator chains so DS latency hides behind WMMA.
// ---------------------------------------------------------------------------
__global__ __launch_bounds__(128) void attn_causal(
    const __bf16* __restrict__ Q,   // [B*S, H*DK]
    const __bf16* __restrict__ Kb,  // [B*S, HKV*DK]
    const __bf16* __restrict__ Vb,  // [B*S, HKV*DK]
    __bf16* __restrict__ Ob) {      // [B*S, H*DK]
  __shared__ alignas(16) __bf16 sK[2][64 * 128];
  __shared__ alignas(16) __bf16 sV[2][64 * 128];
  __shared__ alignas(16) __bf16 sP[4][16 * 64];

  const int bh   = blockIdx.y;
  const int b    = bh / kH;
  const int h    = bh % kH;
  const int kvh  = h / kGrp;
  const int qt   = blockIdx.x;  // 64-row q tile
  const int tid  = threadIdx.x;
  const int wave = tid >> 5;
  const int lane = tid & 31;
  const int half = lane >> 4;
  const int ln   = lane & 15;
  const int q0   = qt * 64 + wave * 16;  // wave's first q row

  // Q fragments resident in registers: 16 rows x 128 dk = 4 k-steps
  v16bf qf[4];
  {
    const __bf16* qbase = Q + (size_t)(b * kS + q0) * kD + h * kDK;
#pragma unroll
    for (int ks = 0; ks < 4; ++ks) qf[ks] = ld_frag_a(qbase + ks * 32, kD);
  }

  v8f of[8];
#pragma unroll
  for (int nt = 0; nt < 8; ++nt) of[nt] = {};
  float m[8], l[8];
#pragma unroll
  for (int jj = 0; jj < 8; ++jj) { m[jj] = -3.0e38f; l[jj] = 0.f; }

  const float scale = 0.08838834764831845f;  // 1/sqrt(128)

  // per-thread KV tile-load geometry: 64 bf16 per tile per thread = 8 b128
  const int r  = tid >> 1;
  const int c0 = (tid & 1) << 6;

  auto issue_kv = [&](int j, int buf) {
    const size_t grow = (size_t)(b * kS + j * 64 + r) * kKVW + kvh * kDK + c0;
    const uint64_t gk = (uint64_t)(uintptr_t)(Kb + grow);
    const uint64_t gv = (uint64_t)(uintptr_t)(Vb + grow);
    const uint32_t lk = lds_off(&sK[buf][r * 128 + c0]);
    const uint32_t lv = lds_off(&sV[buf][r * 128 + c0]);
    async_b128<0>(lk, gk);   async_b128<16>(lk, gk);
    async_b128<32>(lk, gk);  async_b128<48>(lk, gk);
    async_b128<64>(lk, gk);  async_b128<80>(lk, gk);
    async_b128<96>(lk, gk);  async_b128<112>(lk, gk);
    async_b128<0>(lv, gv);   async_b128<16>(lv, gv);
    async_b128<32>(lv, gv);  async_b128<48>(lv, gv);
    async_b128<64>(lv, gv);  async_b128<80>(lv, gv);
    async_b128<96>(lv, gv);  async_b128<112>(lv, gv);
  };

  issue_kv(0, 0);
  int cur = 0;
  for (int j = 0; j <= qt; ++j) {
    wait_async0();
    __syncthreads();
    if (j < qt) issue_kv(j + 1, cur ^ 1);

    // scores: 16(q) x 64(kv); two independent accumulator chains in flight
    v8f sf[4];
#pragma unroll
    for (int np = 0; np < 2; ++np) {
      v8f a0 = {}, a1 = {};
#pragma unroll
      for (int ks = 0; ks < 4; ++ks) {
        v16bf kf0, kf1;
#pragma unroll
        for (int i = 0; i < 8; ++i) {
          const int kb   = (half << 4) + (i << 1);
          const int row0 = (np * 32 + ln) * 128 + ks * 32 + kb;
          const int row1 = row0 + 16 * 128;
          kf0[2 * i]     = sK[cur][row0];
          kf0[2 * i + 1] = sK[cur][row0 + 1];
          kf1[2 * i]     = sK[cur][row1];
          kf1[2 * i + 1] = sK[cur][row1 + 1];
        }
        a0 = wmma_bf16(qf[ks], kf0, a0);
        a1 = wmma_bf16(qf[ks], kf1, a1);
      }
      sf[np * 2]     = a0;
      sf[np * 2 + 1] = a1;
    }

    // scale + causal mask (diagonal tile only)
    const bool diag = (j == qt);
#pragma unroll
    for (int nt = 0; nt < 4; ++nt)
#pragma unroll
      for (int jj = 0; jj < 8; ++jj) {
        float v = sf[nt][jj] * scale;
        if (diag) {
          const int qrel = wave * 16 + jj + 8 * half;
          const int krel = nt * 16 + ln;
          if (krel > qrel) v = -3.0e38f;
        }
        sf[nt][jj] = v;
      }

    // online softmax: row reductions across the 16 lanes of each half
    float corr[8];
#pragma unroll
    for (int jj = 0; jj < 8; ++jj) {
      float rm = fmaxf(fmaxf(sf[0][jj], sf[1][jj]), fmaxf(sf[2][jj], sf[3][jj]));
      rm = fmaxf(rm, __shfl_xor(rm, 1));
      rm = fmaxf(rm, __shfl_xor(rm, 2));
      rm = fmaxf(rm, __shfl_xor(rm, 4));
      rm = fmaxf(rm, __shfl_xor(rm, 8));
      const float mn = fmaxf(m[jj], rm);
      corr[jj] = __expf(m[jj] - mn);
      m[jj]    = mn;
      float rs = 0.f;
#pragma unroll
      for (int nt = 0; nt < 4; ++nt) {
        const float p = __expf(sf[nt][jj] - mn);
        sf[nt][jj]    = p;
        rs += p;
      }
      rs += __shfl_xor(rs, 1);
      rs += __shfl_xor(rs, 2);
      rs += __shfl_xor(rs, 4);
      rs += __shfl_xor(rs, 8);
      l[jj] = l[jj] * corr[jj] + rs;
    }

    // P -> LDS (bf16) so it can be re-fragmented as an A operand
#pragma unroll
    for (int nt = 0; nt < 4; ++nt)
#pragma unroll
      for (int jj = 0; jj < 8; ++jj)
        sP[wave][(jj + 8 * half) * 64 + nt * 16 + ln] = (__bf16)sf[nt][jj];
    asm volatile("s_wait_dscnt 0" ::: "memory");  // in-wave LDS RAW fence

    // rescale running output
#pragma unroll
    for (int nt = 0; nt < 8; ++nt)
#pragma unroll
      for (int jj = 0; jj < 8; ++jj) of[nt][jj] *= corr[jj];

    // out += P @ V  (K-dim = 64 kv -> 2 k-steps; 8 independent chains)
    v16bf pf0 = ld_frag_a(&sP[wave][0], 64);
    v16bf pf1 = ld_frag_a(&sP[wave][0] + 32, 64);
#pragma unroll
    for (int nt = 0; nt < 8; ++nt) {
      v16bf vf0 = ld_frag_b(&sV[cur][0 * 128 + nt * 16], 128);
      v16bf vf1 = ld_frag_b(&sV[cur][32 * 128 + nt * 16], 128);
      of[nt] = wmma_bf16(pf0, vf0, of[nt]);
      of[nt] = wmma_bf16(pf1, vf1, of[nt]);
    }
    cur ^= 1;
  }

  // epilogue: normalize and store bf16 attention output
#pragma unroll
  for (int nt = 0; nt < 8; ++nt)
#pragma unroll
    for (int jj = 0; jj < 8; ++jj) {
      const int qg    = q0 + jj + 8 * half;
      const float val = of[nt][jj] / l[jj];
      Ob[(size_t)(b * kS + qg) * kD + h * kDK + nt * 16 + ln] = (__bf16)val;
    }
}

// ---------------------------------------------------------------------------
// Host launcher
// ---------------------------------------------------------------------------
extern "C" void kernel_launch(void* const* d_in, const int* in_sizes, int n_in,
                              void* d_out, int out_size, void* d_ws,
                              size_t ws_size, hipStream_t stream) {
  (void)in_sizes; (void)n_in; (void)out_size; (void)ws_size;

  const float* x   = (const float*)d_in[0];
  const float* Wq1 = (const float*)d_in[1];
  const float* Wq2 = (const float*)d_in[2];
  const float* Wk1 = (const float*)d_in[3];
  const float* Wk2 = (const float*)d_in[4];
  const float* Wv1 = (const float*)d_in[5];
  const float* Wv2 = (const float*)d_in[6];
  const float* Wo  = (const float*)d_in[7];
  float* out       = (float*)d_out;

  const int MS = kB * kS;  // 8192 token rows

  char* ws   = (char*)d_ws;
  size_t off = 0;
  auto alloc = [&](size_t elems) {
    __bf16* p = (__bf16*)(ws + off);
    off += ((elems * sizeof(__bf16) + 255) / 256) * 256;
    return p;
  };
  __bf16* xb   = alloc((size_t)MS * kD);
  __bf16* wq1b = alloc((size_t)kD * 512);
  __bf16* wq2b = alloc((size_t)512 * kD);
  __bf16* wk1b = alloc((size_t)kD * 512);
  __bf16* wk2b = alloc((size_t)512 * 512);
  __bf16* wv1b = alloc((size_t)kD * 512);
  __bf16* wv2b = alloc((size_t)512 * 512);
  __bf16* wob  = alloc((size_t)kD * kD);
  __bf16* latb = alloc((size_t)MS * 512);   // reused for q/k/v latents
  __bf16* qb   = alloc((size_t)MS * kD);
  __bf16* kbuf = alloc((size_t)MS * kKVW);
  __bf16* vbuf = alloc((size_t)MS * kKVW);
  __bf16* aob  = alloc((size_t)MS * kD);

  auto cvt = [&](const float* s, __bf16* d, int n) {
    cvt_f32_bf16<<<2048, 256, 0, stream>>>(s, d, n);
  };
  cvt(x,   xb,   MS * kD);
  cvt(Wq1, wq1b, kD * 512);
  cvt(Wq2, wq2b, 512 * kD);
  cvt(Wk1, wk1b, kD * 512);
  cvt(Wk2, wk2b, 512 * 512);
  cvt(Wv1, wv1b, kD * 512);
  cvt(Wv2, wv2b, 512 * 512);
  cvt(Wo,  wob,  kD * kD);

  auto gemm_b = [&](const __bf16* A, const __bf16* Bm, __bf16* C, int M, int N,
                    int K) {
    gemm_bf16<__bf16><<<dim3(N / 256, M / 128), 256, 0, stream>>>(A, Bm, C, M, N, K);
  };

  // Q = (x @ Wq1) @ Wq2
  gemm_b(xb, wq1b, latb, MS, 512, kD);
  gemm_b(latb, wq2b, qb, MS, kD, 512);
  // K = (x @ Wk1) @ Wk2
  gemm_b(xb, wk1b, latb, MS, 512, kD);
  gemm_b(latb, wk2b, kbuf, MS, kKVW, 512);
  // V = (x @ Wv1) @ Wv2
  gemm_b(xb, wv1b, latb, MS, 512, kD);
  gemm_b(latb, wv2b, vbuf, MS, kKVW, 512);

  // causal GQA attention
  attn_causal<<<dim3(kS / 64, kB * kH), 128, 0, stream>>>(qb, kbuf, vbuf, aob);

  // final projection, fp32 output
  gemm_bf16<float><<<dim3(kD / 256, MS / 128), 256, 0, stream>>>(
      aob, wob, out, MS, kD, kD);
}
